// GraphLatticeModel_22883585753798
// MI455X (gfx1250) — matrix operand order repaired and checked by reading, not compiled, not run on hardware
//
#include <hip/hip_runtime.h>
#include <hip/hip_bf16.h>
#include <math.h>

#define N_NODES 50000
#define N_EDGES 800000
#define N_GRAPHS 128
#define D_IN 128
#define D_H 256
#define D_L 256
#define D_OUT 32
#define BN_EPS 1e-5f
#define MT_PER_WAVE 5   // M-tiles per wave; (N_NODES/16) = 3125 is divisible by 5

typedef __attribute__((ext_vector_type(16))) __bf16 v16bf;
typedef __attribute__((ext_vector_type(8)))  __bf16 v8bf;
typedef __attribute__((ext_vector_type(8)))  float  v8f;

// ---------------- elementwise helpers ----------------
__global__ void k_fill_zero(float* __restrict__ p, int n) {
  int i = blockIdx.x * blockDim.x + threadIdx.x;
  if (i < n) p[i] = 0.0f;
}

// convert fp32 W[K x N] -> bf16 WT[N x K] (transposed, fragment-friendly)
__global__ void k_cvt_bf16_t(const float* __restrict__ in, __bf16* __restrict__ out,
                             int K, int N) {
  int i = blockIdx.x * blockDim.x + threadIdx.x;
  if (i >= K * N) return;
  int k = i / N, n = i - k * N;
  out[(long long)n * K + k] = (__bf16)in[i];
}

__global__ void k_add_cvt(const float* __restrict__ x, const float* __restrict__ agg,
                          __bf16* __restrict__ out, int n) {
  int i = blockIdx.x * blockDim.x + threadIdx.x;
  if (i < n) out[i] = (__bf16)(x[i] + agg[i]);
}

// ---------------- edge aggregation: agg[dst] += x[src] ----------------
__global__ void k_scatter_add(const float* __restrict__ x, const int* __restrict__ src,
                              const int* __restrict__ dst, float* __restrict__ agg,
                              int ne, int D) {
  long long i = (long long)blockIdx.x * blockDim.x + threadIdx.x;
  long long tot = (long long)ne * D;
  if (i >= tot) return;
  int e = (int)(i / D);
  int f = (int)(i - (long long)e * D);
  int s = src[e], d = dst[e];
  atomicAdd(&agg[(long long)d * D + f], x[(long long)s * D + f]);
}

// ---------------- BF16 WMMA GEMM ----------------
// C = act(A[MxK] * B[KxN] + bias); B passed pre-transposed as BT[NxK].
// One wave computes an 80x16 strip of C (5 M-tiles), reusing each B fragment
// across 5 back-to-back v_wmma_f32_16x16x32_bf16 on independent accumulators.
__global__ void k_wmma_gemm(const __bf16* __restrict__ A, const __bf16* __restrict__ BT,
                            const float* __restrict__ bias,
                            float* __restrict__ outF, __bf16* __restrict__ outB,
                            int M, int N, int K, int relu) {
  int wid  = blockIdx.x * (blockDim.x >> 5) + ((int)threadIdx.x >> 5);
  int lane = threadIdx.x & 31;
  int tn = N >> 4;
  int mgroups = (M >> 4) / MT_PER_WAVE;
  int total = mgroups * tn;
  if (wid >= total) return;                 // wave-uniform exit: EXEC stays all-1s
  int mg = wid / tn, nt = wid - mg * tn;
  int m0 = mg * (16 * MT_PER_WAVE), n0 = nt << 4;
  int l16 = lane & 15;                      // A row / B,C column within tile
  int kh  = lane >> 4;                      // lane half

  v8f acc[MT_PER_WAVE];
#pragma unroll
  for (int t = 0; t < MT_PER_WAVE; ++t) acc[t] = (v8f){};

  // A: per lane one contiguous 16-elem run at K = kh*16 (two 16B loads)
  const __bf16* Ar = A + (long long)(m0 + l16) * K + (kh << 4);
  const long long rowStride = (long long)16 * K;
  // BT: per lane one contiguous 16-elem (32B) run at K = kh*16
  const __bf16* Bt = BT + (long long)(n0 + l16) * K + (kh << 4);

  for (int k0 = 0; k0 < K; k0 += 32) {
    v16bf b = *(const v16bf*)(Bt + k0);     // 32B aligned: two b128 loads
    v16bf a[MT_PER_WAVE];
#pragma unroll
    for (int t = 0; t < MT_PER_WAVE; ++t) {
      const __bf16* Ap = Ar + (long long)t * rowStride + k0;
      v8bf lo = *(const v8bf*)Ap;
      v8bf hi = *(const v8bf*)(Ap + 8);
      a[t] = __builtin_shufflevector(lo, hi, 0,1,2,3,4,5,6,7,8,9,10,11,12,13,14,15);
    }
#pragma unroll
    for (int t = 0; t < MT_PER_WAVE; ++t)
      acc[t] = __builtin_amdgcn_wmma_f32_16x16x32_bf16(false, a[t], false, b,
                                                       (short)0, acc[t], false, false);
  }

  float bv = bias[n0 + l16];
#pragma unroll
  for (int t = 0; t < MT_PER_WAVE; ++t) {
#pragma unroll
    for (int r = 0; r < 8; ++r) {
      float v = acc[t][r] + bv;
      if (relu) v = fmaxf(v, 0.0f);
      long long o = (long long)(m0 + t * 16 + r + (kh << 3)) * N + (n0 + l16);
      if (outF) outF[o] = v;
      if (outB) outB[o] = (__bf16)v;
    }
  }
}

// ---------------- batchnorm ----------------
__global__ void k_bn_stats(const float* __restrict__ h, float* __restrict__ stats,
                           int M, int N) {
  __shared__ float s1[256], s2[256];
  int col = blockIdx.x;
  float a = 0.f, b = 0.f;
  for (int r = threadIdx.x; r < M; r += blockDim.x) {
    float v = h[(long long)r * N + col];
    a += v; b += v * v;
  }
  s1[threadIdx.x] = a; s2[threadIdx.x] = b;
  __syncthreads();
  for (int o = 128; o > 0; o >>= 1) {
    if ((int)threadIdx.x < o) {
      s1[threadIdx.x] += s1[threadIdx.x + o];
      s2[threadIdx.x] += s2[threadIdx.x + o];
    }
    __syncthreads();
  }
  if (threadIdx.x == 0) {
    float mu = s1[0] / (float)M;
    stats[col]     = mu;
    stats[N + col] = s2[0] / (float)M - mu * mu;
  }
}

__global__ void k_bn_apply(const float* __restrict__ h, const float* __restrict__ stats,
                           const float* __restrict__ gamma, const float* __restrict__ beta,
                           float* __restrict__ out, int M, int N) {
  long long i = (long long)blockIdx.x * blockDim.x + threadIdx.x;
  if (i >= (long long)M * N) return;
  int c = (int)(i % N);
  float mu = stats[c], var = stats[N + c];
  float v = gamma[c] * (h[i] - mu) * rsqrtf(var + BN_EPS) + beta[c];
  out[i] = fmaxf(v, 0.0f);
}

// ---------------- global mean pool ----------------
__global__ void k_pool(const float* __restrict__ x, const int* __restrict__ batch,
                       float* __restrict__ pool, float* __restrict__ cnt, int M, int N) {
  long long i = (long long)blockIdx.x * blockDim.x + threadIdx.x;
  if (i >= (long long)M * N) return;
  int r = (int)(i / N), c = (int)(i - (long long)r * N);
  int g = batch[r];
  atomicAdd(&pool[(long long)g * N + c], x[i]);
  if (c == 0) atomicAdd(&cnt[g], 1.0f);
}

__global__ void k_comb(const float* __restrict__ pool, const float* __restrict__ cnt,
                       const float* __restrict__ lat, float* __restrict__ comb,
                       int G, int D) {
  int i = blockIdx.x * blockDim.x + threadIdx.x;
  if (i >= G * 2 * D) return;
  int g = i / (2 * D), j = i - g * 2 * D;
  comb[i] = (j < D) ? pool[g * D + j] / fmaxf(cnt[g], 1.0f)
                    : lat[g * D + (j - D)];
}

// ---------------- small dense layers (head) ----------------
// mode: 0 = linear, 1 = relu, 2 = tanh*0.5+0.5
__global__ void k_dense(const float* __restrict__ in, const float* __restrict__ w,
                        const float* __restrict__ b, float* __restrict__ out,
                        int M, int K, int N, int mode) {
  int i = blockIdx.x * blockDim.x + threadIdx.x;
  if (i >= M * N) return;
  int m = i / N, n = i - m * N;
  float acc = b[n];
  const float* ir = in + (long long)m * K;
  for (int k = 0; k < K; ++k) acc += ir[k] * w[(long long)k * N + n];
  if (mode == 1) acc = fmaxf(acc, 0.0f);
  else if (mode == 2) acc = tanhf(acc) * 0.5f + 0.5f;
  out[i] = acc;
}

// ---------------- driver ----------------
extern "C" void kernel_launch(void* const* d_in, const int* in_sizes, int n_in,
                              void* d_out, int out_size, void* d_ws, size_t ws_size,
                              hipStream_t stream) {
  (void)in_sizes; (void)n_in; (void)out_size; (void)ws_size;

  const float* x0    = (const float*)d_in[0];
  const int*   ei    = (const int*)d_in[1];
  const int*   batch = (const int*)d_in[2];
  const float* lat   = (const float*)d_in[3];
  const float* Wa[3] = {(const float*)d_in[4],  (const float*)d_in[10], (const float*)d_in[16]};
  const float* Ba[3] = {(const float*)d_in[5],  (const float*)d_in[11], (const float*)d_in[17]};
  const float* Wb[3] = {(const float*)d_in[6],  (const float*)d_in[12], (const float*)d_in[18]};
  const float* Bb[3] = {(const float*)d_in[7],  (const float*)d_in[13], (const float*)d_in[19]};
  const float* Ga[3] = {(const float*)d_in[8],  (const float*)d_in[14], (const float*)d_in[20]};
  const float* Be[3] = {(const float*)d_in[9],  (const float*)d_in[15], (const float*)d_in[21]};
  const float* lw1 = (const float*)d_in[22]; const float* lb1 = (const float*)d_in[23];
  const float* lw2 = (const float*)d_in[24]; const float* lb2 = (const float*)d_in[25];
  const float* f1w = (const float*)d_in[26]; const float* f1b = (const float*)d_in[27];
  const float* f2w = (const float*)d_in[28]; const float* f2b = (const float*)d_in[29];
  float* out = (float*)d_out;

  // carve workspace (256B aligned slices)
  char* wsb = (char*)d_ws;
  size_t off = 0;
  auto carve = [&](size_t bytes) -> void* {
    void* p = wsb + off;
    off = (off + bytes + 255) & ~(size_t)255;
    return p;
  };
  float*  X    = (float*)carve(sizeof(float) * (size_t)N_NODES * D_H);
  float*  AGG  = (float*)carve(sizeof(float) * (size_t)N_NODES * D_H);  // reused as H
  __bf16* HB   = (__bf16*)carve(sizeof(__bf16) * (size_t)N_NODES * D_H);
  __bf16* ZB   = (__bf16*)carve(sizeof(__bf16) * (size_t)N_NODES * D_H);
  __bf16* W1T  = (__bf16*)carve(sizeof(__bf16) * D_H * D_H);  // [N x K] transposed
  __bf16* W2T  = (__bf16*)carve(sizeof(__bf16) * D_H * D_H);  // [N x K] transposed
  float*  STAT = (float*)carve(sizeof(float) * 2 * D_H);
  float*  POOL = (float*)carve(sizeof(float) * N_GRAPHS * D_H);
  float*  CNT  = (float*)carve(sizeof(float) * N_GRAPHS);
  float*  LAT1 = (float*)carve(sizeof(float) * N_GRAPHS * D_L);
  float*  LAT2 = (float*)carve(sizeof(float) * N_GRAPHS * D_L);
  float*  COMB = (float*)carve(sizeof(float) * N_GRAPHS * (D_H + D_L));
  float*  FC1  = (float*)carve(sizeof(float) * N_GRAPHS * D_H);

  const int* src = ei;
  const int* dst = ei + N_EDGES;
  const int TB = 256;
  auto nb = [](long long n, int tb) { return (unsigned)((n + tb - 1) / tb); };

  const float* xin = x0;
  for (int l = 0; l < 3; ++l) {
    int Din = (l == 0) ? D_IN : D_H;
    long long nElem = (long long)N_NODES * Din;

    k_fill_zero  <<<nb(nElem, TB), TB, 0, stream>>>(AGG, (int)nElem);
    k_scatter_add<<<nb((long long)N_EDGES * Din, TB), TB, 0, stream>>>(xin, src, dst, AGG, N_EDGES, Din);
    k_add_cvt    <<<nb(nElem, TB), TB, 0, stream>>>(xin, AGG, HB, (int)nElem);
    // W1 is [Din x D_H] -> W1T [D_H x Din]; W2 is [D_H x D_H] -> W2T [D_H x D_H]
    k_cvt_bf16_t <<<nb((long long)Din * D_H, TB), TB, 0, stream>>>(Wa[l], W1T, Din, D_H);
    k_cvt_bf16_t <<<nb((long long)D_H * D_H, TB), TB, 0, stream>>>(Wb[l], W2T, D_H, D_H);

    int waves = ((N_NODES / 16) / MT_PER_WAVE) * (D_H / 16);  // 625 * 16 = 10000
    unsigned gblocks = (unsigned)((waves + 7) / 8);
    // Z = relu(H @ W1 + b1) -> bf16
    k_wmma_gemm<<<gblocks, TB, 0, stream>>>(HB, W1T, Ba[l], nullptr, ZB, N_NODES, D_H, Din, 1);
    // H = Z @ W2 + b2 -> fp32 (into AGG, now free)
    k_wmma_gemm<<<gblocks, TB, 0, stream>>>(ZB, W2T, Bb[l], AGG, nullptr, N_NODES, D_H, D_H, 0);

    k_bn_stats<<<D_H, 256, 0, stream>>>(AGG, STAT, N_NODES, D_H);
    k_bn_apply<<<nb((long long)N_NODES * D_H, TB), TB, 0, stream>>>(AGG, STAT, Ga[l], Be[l], X, N_NODES, D_H);
    xin = X;
  }

  // global mean pool
  k_fill_zero<<<nb(N_GRAPHS * D_H, TB), TB, 0, stream>>>(POOL, N_GRAPHS * D_H);
  k_fill_zero<<<1, TB, 0, stream>>>(CNT, N_GRAPHS);
  k_pool<<<nb((long long)N_NODES * D_H, TB), TB, 0, stream>>>(X, batch, POOL, CNT, N_NODES, D_H);

  // lattice MLP + head
  k_dense<<<nb(N_GRAPHS * D_L, TB), TB, 0, stream>>>(lat,  lw1, lb1, LAT1, N_GRAPHS, 9,   D_L, 1);
  k_dense<<<nb(N_GRAPHS * D_L, TB), TB, 0, stream>>>(LAT1, lw2, lb2, LAT2, N_GRAPHS, D_L, D_L, 1);
  k_comb <<<nb(N_GRAPHS * (D_H + D_L), TB), TB, 0, stream>>>(POOL, CNT, LAT2, COMB, N_GRAPHS, D_H);
  k_dense<<<nb(N_GRAPHS * D_H, TB), TB, 0, stream>>>(COMB, f1w, f1b, FC1, N_GRAPHS, D_H + D_L, D_H, 1);
  k_dense<<<nb(N_GRAPHS * D_OUT, TB), TB, 0, stream>>>(FC1, f2w, f2b, out, N_GRAPHS, D_H, D_OUT, 2);
}
